// Net_56006373540327
// MI455X (gfx1250) — matrix operand order repaired and checked by reading, not compiled, not run on hardware
//
#include <hip/hip_runtime.h>
#include <math.h>

#define NN 50000
#define NE 1600000
#define MT (NN/16)   // 3125 row tiles (50000 divisible by 16)
#define CDIV(a,b) (((a)+(b)-1)/(b))

typedef __attribute__((ext_vector_type(16))) _Float16 v16h;
typedef __attribute__((ext_vector_type(8)))  _Float16 v8h;
typedef __attribute__((ext_vector_type(8)))  float    v8f;
typedef __attribute__((ext_vector_type(4)))  float    v4f;

// ---- monotone float<->uint mapping for atomicMax on signed floats ----
__device__ __forceinline__ unsigned fenc(float f) {
    unsigned b = __float_as_uint(f);
    return (b & 0x80000000u) ? ~b : (b | 0x80000000u);
}
__device__ __forceinline__ float fdec(unsigned u) {
    unsigned b = (u & 0x80000000u) ? (u & 0x7fffffffu) : ~u;
    return __uint_as_float(b);
}

// ---------------- packing / prep kernels ----------------

// pad node features [N,12] -> [N,32] (zeros beyond col 12)
__global__ void pack_x_k(const float* __restrict__ x, float* __restrict__ hpad) {
    int g = blockIdx.x * blockDim.x + threadIdx.x;
    if (g >= NN * 32) return;
    int n = g >> 5, c = g & 31;
    hpad[g] = (c < 12) ? x[n * 12 + c] : 0.f;
}

// W float [Ko][NcolsSrc] row-major -> half Bt[(nOff+n)][Kp], zero pad K
__global__ void pack_bt_k(const float* __restrict__ W, int Ko, int NcolsSrc,
                          _Float16* __restrict__ out, int nOff, int Kp, int ncols) {
    int g = blockIdx.x * blockDim.x + threadIdx.x;
    if (g >= ncols * Kp) return;
    int n = g / Kp, k = g % Kp;
    float v = (k < Ko) ? W[k * NcolsSrc + n] : 0.f;
    out[(size_t)(nOff + n) * Kp + k] = (_Float16)v;
}

__global__ void copy_f_k(float* __restrict__ dst, const float* __restrict__ src, int n) {
    int g = blockIdx.x * blockDim.x + threadIdx.x;
    if (g < n) dst[g] = src[g];
}

// fold BN into scale/shift: y = x*scale + shift
__global__ void bn_prep_k(const float* __restrict__ g_, const float* __restrict__ b_,
                          const float* __restrict__ m_, const float* __restrict__ v_,
                          float* __restrict__ out) {
    int c = threadIdx.x;
    if (c >= 32) return;
    float sc = g_[c] * rsqrtf(v_[c] + 1e-5f);
    out[c] = sc;
    out[32 + c] = b_[c] - m_[c] * sc;
}

__global__ void init_stats_k(unsigned* __restrict__ nmax, float* __restrict__ nsum) {
    int g = blockIdx.x * blockDim.x + threadIdx.x;
    if (g >= 2 * NN) return;
    nmax[g] = 0u;       // encodes below every real float
    nsum[g] = 0.f;
}

// ---------------- WMMA GEMM ----------------
// Out[M,Nc] = epilogue( A_f32[M,K](lda) x Bt_f16[Nc][Kp] + bias )
// epil: 0 = +bias, 1 = relu(+bias), 2 = relu(+bias)*scale+shift (folded BN)
// One wave per 16x16 tile; 4 waves/block stacked along M.
__global__ __launch_bounds__(128)
void wmma_gemm_k(const float* __restrict__ A, int lda,
                 const _Float16* __restrict__ Bt, int Kp,
                 const float* __restrict__ bias,
                 float* __restrict__ Out, int ldo,
                 int Mtiles, int nK, int epil,
                 const float* __restrict__ scale, const float* __restrict__ shift) {
    int wave = threadIdx.x >> 5;
    int tileM = blockIdx.x * 4 + wave;
    if (tileM >= Mtiles) return;                 // wave-uniform guard
    int tileN = blockIdx.y;
    int lane = threadIdx.x & 31;
    int h0 = lane >> 4;                          // half-wave id
    int lm = lane & 15;

    int row = tileM * 16 + lm;                   // A row for this lane
    int n   = tileN * 16 + lm;                   // B column / output column
    const float*    arow = A  + (size_t)row * lda;
    const _Float16* brow = Bt + (size_t)n   * Kp;

    v8f c = {};
    for (int kc = 0; kc < nK; ++kc) {
        // A 16-bit layout: elems 0..7 -> K = kc*32 + 8*h0 + e
        //                  elems 8..15 -> K = kc*32 + 16 + 8*h0 + e
        int ka = kc * 32 + 8 * h0;
        const v4f* ap0 = (const v4f*)(arow + ka);
        const v4f* ap1 = (const v4f*)(arow + ka + 16);
        v4f a0 = ap0[0], a1 = ap0[1], a2 = ap1[0], a3 = ap1[1];
        v16h a;
        #pragma unroll
        for (int e = 0; e < 4; ++e) {
            a[e]      = (_Float16)a0[e];
            a[4 + e]  = (_Float16)a1[e];
            a[8 + e]  = (_Float16)a2[e];
            a[12 + e] = (_Float16)a3[e];
        }
        // B 16-bit layout: elems 0..7 -> K = kc*32 + 8*h0 + e, elems 8..15 -> K = kc*32 + 16 + 8*h0 + e
        const v8h* bp = (const v8h*)(brow + kc * 32 + 8 * h0);
        v8h b0 = bp[0], b1 = bp[2];
        v16h b;
        #pragma unroll
        for (int e = 0; e < 8; ++e) { b[e] = b0[e]; b[8 + e] = b1[e]; }

        c = __builtin_amdgcn_wmma_f32_16x16x32_f16(false, a, false, b,
                                                   (short)0, c, false, false);
    }

    float bn = bias[n];
    float sc = (epil == 2) ? scale[n] : 1.f;
    float sh = (epil == 2) ? shift[n] : 0.f;
    #pragma unroll
    for (int r = 0; r < 8; ++r) {
        int m = tileM * 16 + r + 8 * h0;         // D: VGPR r, half-wave h0 -> M = r + 8*h0
        float v = c[r] + bn;
        if (epil >= 1) v = v > 0.f ? v : 0.f;
        v = v * sc + sh;
        Out[(size_t)m * ldo + n] = v;
    }
}

// ---------------- edge-stage kernels ----------------
// qkvs row layout: [q(64) | k(64) | v(64) | s(64)] per node, ld = 256

__global__ void edge_alpha_k(const int* __restrict__ ei, const float* __restrict__ ea,
                             const float* __restrict__ qkvs, const float* __restrict__ We,
                             float* __restrict__ alpha, unsigned* __restrict__ nmax) {
    int i = blockIdx.x * blockDim.x + threadIdx.x;
    if (i >= NE) return;
    int s = ei[i], d = ei[NE + i];
    float ea0 = ea[2 * (size_t)i], ea1 = ea[2 * (size_t)i + 1];
    #pragma unroll
    for (int h = 0; h < 2; ++h) {
        const v4f* qp = (const v4f*)(qkvs + (size_t)d * 256 + h * 32);
        const v4f* kp = (const v4f*)(qkvs + (size_t)s * 256 + 64 + h * 32);
        const v4f* w0 = (const v4f*)(We + h * 32);
        const v4f* w1 = (const v4f*)(We + 64 + h * 32);
        float acc = 0.f;
        #pragma unroll
        for (int j = 0; j < 8; ++j) {
            v4f q = qp[j], k = kp[j];
            v4f e = ea0 * w0[j] + ea1 * w1[j];
            v4f t = q * (k + e);
            acc += t[0] + t[1] + t[2] + t[3];
        }
        float a = acc * 0.17677669529663687f;    // 1/sqrt(32)
        alpha[2 * (size_t)i + h] = a;
        atomicMax(&nmax[2 * d + h], fenc(a));
    }
}

__global__ void edge_exp_k(const int* __restrict__ ei, float* __restrict__ alpha,
                           const unsigned* __restrict__ nmax, float* __restrict__ nsum) {
    int g = blockIdx.x * blockDim.x + threadIdx.x;
    if (g >= 2 * NE) return;
    int i = g >> 1, h = g & 1;
    int d = ei[NE + i];
    float m = fdec(nmax[2 * d + h]);
    float ex = expf(alpha[g] - m);
    alpha[g] = ex;
    atomicAdd(&nsum[2 * d + h], ex);
}

// one thread per (edge, out-channel-of-64); accumulates into the S slice
__global__ void edge_scatter_k(const int* __restrict__ ei, const float* __restrict__ ea,
                               float* __restrict__ qkvs, const float* __restrict__ We,
                               const float* __restrict__ alpha, const float* __restrict__ nsum) {
    long long g = (long long)blockIdx.x * blockDim.x + threadIdx.x;
    if (g >= (long long)NE * 64) return;
    int i = (int)(g >> 6);
    int c = (int)(g & 63);
    int h = c >> 5;
    int s = ei[i], d = ei[NE + i];
    float ex  = alpha[2 * (size_t)i + h];
    float den = nsum[2 * d + h] + 1e-16f;
    float a   = ex / den;
    float e   = ea[2 * (size_t)i] * We[c] + ea[2 * (size_t)i + 1] * We[64 + c];
    float val = a * (qkvs[(size_t)s * 256 + 128 + c] + e);
    atomicAdd(&qkvs[(size_t)d * 256 + 192 + c], val);
}

// final 32->1 dot + sigmoid
__global__ void out_final_k(const float* __restrict__ h, const float* __restrict__ W,
                            const float* __restrict__ b, float* __restrict__ out) {
    int n = blockIdx.x * blockDim.x + threadIdx.x;
    if (n >= NN) return;
    const v4f* hp = (const v4f*)(h + (size_t)n * 32);
    const v4f* wp = (const v4f*)W;
    float acc = b[0];
    #pragma unroll
    for (int j = 0; j < 8; ++j) {
        v4f t = hp[j] * wp[j];
        acc += t[0] + t[1] + t[2] + t[3];
    }
    out[n] = 1.f / (1.f + expf(-acc));
}

// ---------------- orchestration ----------------
extern "C" void kernel_launch(void* const* d_in, const int* in_sizes, int n_in,
                              void* d_out, int out_size, void* d_ws, size_t ws_size,
                              hipStream_t stream) {
    (void)in_sizes; (void)n_in; (void)out_size; (void)ws_size;
    const float* x     = (const float*)d_in[0];
    const int*   ei    = (const int*)  d_in[1];
    const float* ea    = (const float*)d_in[2];
    const float* c0_Wq = (const float*)d_in[3];
    const float* c0_bq = (const float*)d_in[4];
    const float* c0_Wk = (const float*)d_in[5];
    const float* c0_bk = (const float*)d_in[6];
    const float* c0_Wv = (const float*)d_in[7];
    const float* c0_bv = (const float*)d_in[8];
    const float* c0_We = (const float*)d_in[9];
    const float* c0_Ws = (const float*)d_in[10];
    const float* c0_bs = (const float*)d_in[11];
    const float* t0_W  = (const float*)d_in[12];
    const float* t0_b  = (const float*)d_in[13];
    const float* bn0_g = (const float*)d_in[14];
    const float* bn0_b = (const float*)d_in[15];
    const float* bn0_m = (const float*)d_in[16];
    const float* bn0_v = (const float*)d_in[17];
    const float* cWq   = (const float*)d_in[18];
    const float* cbq   = (const float*)d_in[19];
    const float* cWk   = (const float*)d_in[20];
    const float* cbk   = (const float*)d_in[21];
    const float* cWv   = (const float*)d_in[22];
    const float* cbv   = (const float*)d_in[23];
    const float* cWe   = (const float*)d_in[24];
    const float* cWs   = (const float*)d_in[25];
    const float* cbs   = (const float*)d_in[26];
    const float* tW    = (const float*)d_in[27];
    const float* tb    = (const float*)d_in[28];
    const float* bng   = (const float*)d_in[29];
    const float* bnb   = (const float*)d_in[30];
    const float* bnm   = (const float*)d_in[31];
    const float* bnv   = (const float*)d_in[32];
    const float* o1_W  = (const float*)d_in[33];
    const float* o1_b  = (const float*)d_in[34];
    const float* o2_W  = (const float*)d_in[35];
    const float* o2_b  = (const float*)d_in[36];
    const float* o3_W  = (const float*)d_in[37];
    const float* o3_b  = (const float*)d_in[38];

    // workspace carve (256B aligned chunks)
    char* w = (char*)d_ws;
    size_t off = 0;
    auto take = [&](size_t bytes) -> char* {
        char* p = w + off;
        off += (bytes + 255) & ~(size_t)255;
        return p;
    };
    float*    hpad    = (float*)take((size_t)NN * 32 * 4);   // padded x, later reused as MLP h3
    float*    qkvs    = (float*)take((size_t)NN * 256 * 4);  // q|k|v|s per node
    float*    alpha   = (float*)take((size_t)NE * 2 * 4);
    unsigned* nmax    = (unsigned*)take((size_t)NN * 2 * 4);
    float*    nsum    = (float*)take((size_t)NN * 2 * 4);
    float*    h2      = (float*)take((size_t)NN * 64 * 4);
    float*    hfeat   = (float*)take((size_t)NN * 32 * 4);
    _Float16* wconv   = (_Float16*)take((size_t)5 * 256 * 32 * 2);
    _Float16* wt      = (_Float16*)take((size_t)5 * 32 * 64 * 2);
    _Float16* wo1     = (_Float16*)take((size_t)64 * 32 * 2);
    _Float16* wo2     = (_Float16*)take((size_t)32 * 64 * 2);
    float*    bias256 = (float*)take((size_t)5 * 256 * 4);
    float*    bnss    = (float*)take((size_t)5 * 64 * 4);    // per layer: scale[32] | shift[32]

    const int T = 256;

    // --- weight / input prep ---
    pack_x_k<<<CDIV(NN * 32, T), T, 0, stream>>>(x, hpad);

    pack_bt_k<<<CDIV(2048, T), T, 0, stream>>>(c0_Wq, 12, 64, wconv, 0,   32, 64);
    pack_bt_k<<<CDIV(2048, T), T, 0, stream>>>(c0_Wk, 12, 64, wconv, 64,  32, 64);
    pack_bt_k<<<CDIV(2048, T), T, 0, stream>>>(c0_Wv, 12, 64, wconv, 128, 32, 64);
    pack_bt_k<<<CDIV(2048, T), T, 0, stream>>>(c0_Ws, 12, 64, wconv, 192, 32, 64);
    copy_f_k<<<1, 64, 0, stream>>>(bias256 + 0,   c0_bq, 64);
    copy_f_k<<<1, 64, 0, stream>>>(bias256 + 64,  c0_bk, 64);
    copy_f_k<<<1, 64, 0, stream>>>(bias256 + 128, c0_bv, 64);
    copy_f_k<<<1, 64, 0, stream>>>(bias256 + 192, c0_bs, 64);
    pack_bt_k<<<CDIV(2048, T), T, 0, stream>>>(t0_W, 64, 32, wt, 0, 64, 32);
    bn_prep_k<<<1, 32, 0, stream>>>(bn0_g, bn0_b, bn0_m, bn0_v, bnss);

    for (int l = 1; l < 5; ++l) {
        int j = l - 1;
        pack_bt_k<<<CDIV(2048, T), T, 0, stream>>>(cWq + j * 2048, 32, 64, wconv + l * 8192, 0,   32, 64);
        pack_bt_k<<<CDIV(2048, T), T, 0, stream>>>(cWk + j * 2048, 32, 64, wconv + l * 8192, 64,  32, 64);
        pack_bt_k<<<CDIV(2048, T), T, 0, stream>>>(cWv + j * 2048, 32, 64, wconv + l * 8192, 128, 32, 64);
        pack_bt_k<<<CDIV(2048, T), T, 0, stream>>>(cWs + j * 2048, 32, 64, wconv + l * 8192, 192, 32, 64);
        copy_f_k<<<1, 64, 0, stream>>>(bias256 + l * 256 + 0,   cbq + j * 64, 64);
        copy_f_k<<<1, 64, 0, stream>>>(bias256 + l * 256 + 64,  cbk + j * 64, 64);
        copy_f_k<<<1, 64, 0, stream>>>(bias256 + l * 256 + 128, cbv + j * 64, 64);
        copy_f_k<<<1, 64, 0, stream>>>(bias256 + l * 256 + 192, cbs + j * 64, 64);
        pack_bt_k<<<CDIV(2048, T), T, 0, stream>>>(tW + j * 2048, 64, 32, wt + l * 2048, 0, 64, 32);
        bn_prep_k<<<1, 32, 0, stream>>>(bng + j * 32, bnb + j * 32, bnm + j * 32, bnv + j * 32, bnss + l * 64);
    }
    pack_bt_k<<<CDIV(2048, T), T, 0, stream>>>(o1_W, 32, 64, wo1, 0, 32, 64);
    pack_bt_k<<<CDIV(2048, T), T, 0, stream>>>(o2_W, 64, 32, wo2, 0, 64, 32);

    // --- 5 TransformerConv layers ---
    const float* curA = hpad;
    for (int l = 0; l < 5; ++l) {
        const float* We_l = (l == 0) ? c0_We : (cWe + (l - 1) * 128);
        const float* tb_l = (l == 0) ? t0_b  : (tb  + (l - 1) * 32);

        // fused Q|K|V|S projection: [N,32] x [32,256]  (WMMA)
        dim3 gq(CDIV(MT, 4), 16);
        wmma_gemm_k<<<gq, 128, 0, stream>>>(curA, 32, wconv + l * 8192, 32,
                                            bias256 + l * 256, qkvs, 256, MT, 1,
                                            0, nullptr, nullptr);

        init_stats_k<<<CDIV(2 * NN, T), T, 0, stream>>>(nmax, nsum);
        edge_alpha_k<<<CDIV(NE, T), T, 0, stream>>>(ei, ea, qkvs, We_l, alpha, nmax);
        edge_exp_k<<<CDIV(2 * NE, T), T, 0, stream>>>(ei, alpha, nmax, nsum);
        edge_scatter_k<<<(unsigned)CDIV((long long)NE * 64, T), T, 0, stream>>>(ei, ea, qkvs, We_l, alpha, nsum);

        // t projection + ReLU + BN: [N,64] x [64,32]  (WMMA, A = S slice of qkvs)
        dim3 gt(CDIV(MT, 4), 2);
        wmma_gemm_k<<<gt, 128, 0, stream>>>(qkvs + 192, 256, wt + l * 2048, 64,
                                            tb_l, hfeat, 32, MT, 2,
                                            2, bnss + l * 64, bnss + l * 64 + 32);
        curA = hfeat;
    }

    // --- output MLP (WMMA) ---
    wmma_gemm_k<<<dim3(CDIV(MT, 4), 4), 128, 0, stream>>>(hfeat, 32, wo1, 32, o1_b, h2, 64, MT, 1, 1, nullptr, nullptr);
    wmma_gemm_k<<<dim3(CDIV(MT, 4), 2), 128, 0, stream>>>(h2, 64, wo2, 64, o2_b, hpad, 32, MT, 2, 1, nullptr, nullptr);
    out_final_k<<<CDIV(NN, T), T, 0, stream>>>(hpad, o3_W, o3_b, (float*)d_out);
}
